// Net_70927089926628
// MI455X (gfx1250) — matrix operand order repaired and checked by reading, not compiled, not run on hardware
//
#include <hip/hip_runtime.h>

typedef __attribute__((ext_vector_type(16))) _Float16 v16h;
typedef __attribute__((ext_vector_type(8)))  _Float16 v8h;
typedef __attribute__((ext_vector_type(8)))  float    v8f;
typedef __attribute__((ext_vector_type(4)))  float    v4f;

#define T_STEPS 512
#define BATCH   2048
#define HDIM    128
#define ROWS    16     // batch rows per workgroup
#define NWAVES  8
#define S1_STRIDE 136  // padded row stride (halves): conflict-free ds_load_b128

// 16-lane butterfly add via DPP16 (VALU-only, no LDS pipe).
// Masks {1,2,7,15} span GF(2)^4 -> full 16-lane sum after 4 rounds.
template <int CTRL>
__device__ __forceinline__ float dpp_add(float v) {
  int s = __builtin_amdgcn_update_dpp(0, __float_as_int(v), CTRL, 0xF, 0xF, true);
  return v + __int_as_float(s);
}

__global__ __launch_bounds__(256) void snn_scan_kernel(
    const float* __restrict__ x,
    const float* __restrict__ W_in,   const float* __restrict__ b_in,
    const float* __restrict__ beta_in,const float* __restrict__ thr_in,
    const float* __restrict__ W_h,    const float* __restrict__ b_h,
    const float* __restrict__ beta_h, const float* __restrict__ thr_h,
    const float* __restrict__ W_out,  const float* __restrict__ b_out,
    const float* __restrict__ beta_out,
    float* __restrict__ out)
{
  __shared__ float xtile[T_STEPS * ROWS];                         // 32 KB
  __shared__ __align__(16) _Float16 s1buf[2][ROWS * S1_STRIDE];   // double-buffered spikes
  __shared__ __align__(16) float accbuf[2][ROWS * NWAVES];        // [parity][row*8+wave]

  const int tid   = threadIdx.x;
  const int wv    = tid >> 5;       // wave id 0..7 -> owns N-tile [16*wv, 16*wv+16)
  const int lane  = tid & 31;
  const int li    = lane & 15;
  const int hi    = lane >> 4;      // half-wave
  const int j     = wv * 16 + li;   // layer-2 hidden column owned by this lane
  const int bbase = blockIdx.x * ROWS;

  // layer-1 ownership remap: 2 adjacent columns x 4 rows per lane
  const int cp = li & 7;            // column-pair index within wave
  const int rg = hi * 2 + (li >> 3);// row group 0..3 -> rows 4*rg..4*rg+3
  const int h0 = wv * 16 + 2 * cp;  // first owned layer-1 column (even)

  // stage x[:, bbase:bbase+16] into LDS (read global x exactly once)
  #pragma unroll
  for (int i = 0; i < (T_STEPS * ROWS) / 256; ++i) {
    int flat = tid + 256 * i;
    xtile[flat] = x[(flat >> 4) * BATCH + bbase + (flat & 15)];
  }

  // layer-1 params for the two owned columns
  float p_b1[2], p_t1[2], p_w1[2], p_c1[2];
  #pragma unroll
  for (int c = 0; c < 2; ++c) {
    p_b1[c] = fminf(fmaxf(beta_in[h0 + c], 0.f), 1.f);
    p_t1[c] = thr_in[h0 + c];
    p_w1[c] = W_in[h0 + c];
    p_c1[c] = b_in[h0 + c];
  }
  // layer-2 / output params (one column per lane)
  const float c_b2 = fminf(fmaxf(beta_h[j], 0.f), 1.f);
  const float c_t2 = thr_h[j];
  const float c_c2 = b_h[j];
  const float c_wo = W_out[j];
  const float c_b3 = fminf(fmaxf(beta_out[0], 0.f), 1.f);
  const float c_c3 = b_out[0];

  // W_h held in registers as WMMA B-fragments (B[k][n] = W_h[j, k]).
  // B 32x16 layout: lane -> column N=li (half hi), element e -> K = 32*kt + 16*hi + e
  v16h bfrag[4];
  #pragma unroll
  for (int kt = 0; kt < 4; ++kt) {
    const v4f* src = (const v4f*)(W_h + j * HDIM + kt * 32 + hi * 16);
    #pragma unroll
    for (int q = 0; q < 4; ++q) {
      v4f v = src[q];
      #pragma unroll
      for (int e = 0; e < 4; ++e) bfrag[kt][q * 4 + e] = (_Float16)v[e];
    }
  }

  // persistent bias vector consumed as C of the first WMMA each step
  v8f cinit;
  #pragma unroll
  for (int r = 0; r < 8; ++r) cinit[r] = c_c2;

  // membrane state
  float m1[2][4];                   // [col][row-in-group]
  float m2[8];                      // WMMA C/D layout: reg r <-> row r+8*hi, col j
  #pragma unroll
  for (int r = 0; r < 4; ++r) { m1[0][r] = 0.f; m1[1][r] = 0.f; }
  #pragma unroll
  for (int r = 0; r < 8; ++r) m2[r] = 0.f;
  float m3 = 0.f;                   // meaningful only on tid<16

  __syncthreads();

  for (int t = 0; t < T_STEPS; ++t) {
    const int p = t & 1;

    // ---- phase A: layer-1 leaky + subtract-reset + spike, packed b32 stores ----
    v4f xv = *(const v4f*)&xtile[t * ROWS + rg * 4];   // broadcast within group
    #pragma unroll
    for (int r = 0; r < 4; ++r) {
      union { _Float16 h[2]; unsigned int u; } pk;
      #pragma unroll
      for (int c = 0; c < 2; ++c) {
        float cur = __builtin_fmaf(xv[r], p_w1[c], p_c1[c]);
        float rst = (m1[c][r] > p_t1[c]) ? p_t1[c] : 0.f;  // reset from PREVIOUS mem
        m1[c][r] = __builtin_fmaf(p_b1[c], m1[c][r], cur) - rst;
        pk.h[c] = (m1[c][r] > p_t1[c]) ? (_Float16)1.f : (_Float16)0.f;
      }
      *(unsigned int*)&s1buf[p][(rg * 4 + r) * S1_STRIDE + h0] = pk.u;
    }

    __syncthreads();   // the ONLY barrier per timestep

    // ---- phase B: cur2 = s1 @ W_h^T + b_h via 4 chained WMMAs (K=128) ----
    // A 16x32 f16 layout: row M=li; e0..15 -> K=32kt+16hi+e
    v16h afrag[4];
    #pragma unroll
    for (int kt = 0; kt < 4; ++kt) {
      const _Float16* base = &s1buf[p][li * S1_STRIDE + kt * 32 + hi * 16];
      v8h lo = *(const v8h*)base;
      v8h hh = *(const v8h*)(base + 8);
      #pragma unroll
      for (int e = 0; e < 8; ++e) { afrag[kt][e] = lo[e]; afrag[kt][e + 8] = hh[e]; }
    }
    v8f cacc = __builtin_amdgcn_wmma_f32_16x16x32_f16(
        false, afrag[0], false, bfrag[0], (short)0, cinit, false, false);
    #pragma unroll
    for (int kt = 1; kt < 4; ++kt) {
      cacc = __builtin_amdgcn_wmma_f32_16x16x32_f16(
          false, afrag[kt], false, bfrag[kt], (short)0, cacc, false, false);
    }

    // ---- layer-2 state + spikes + output dot-product partials ----
    float pr[8];
    #pragma unroll
    for (int r = 0; r < 8; ++r) {
      float rst = (m2[r] > c_t2) ? c_t2 : 0.f;
      m2[r] = __builtin_fmaf(c_b2, m2[r], cacc[r]) - rst;
      pr[r] = (m2[r] > c_t2) ? c_wo : 0.f;               // s2 * W_out[j]
    }
    // VALU-only 16-lane reduction: quad_perm xor1, xor2, half-mirror, mirror
    #pragma unroll
    for (int r = 0; r < 8; ++r) {
      pr[r] = dpp_add<0xB1>(pr[r]);    // quad_perm [1,0,3,2]  (xor 1)
      pr[r] = dpp_add<0x4E>(pr[r]);    // quad_perm [2,3,0,1]  (xor 2)
      pr[r] = dpp_add<0x141>(pr[r]);   // ROW_HALF_MIRROR      (xor 7)
      pr[r] = dpp_add<0x140>(pr[r]);   // ROW_MIRROR           (xor 15)
    }
    if (li == 0) {
      #pragma unroll
      for (int r = 0; r < 8; ++r) accbuf[p][(hi * 8 + r) * NWAVES + wv] = pr[r];
    }

    // ---- phase C: output integrator for the PREVIOUS step (acc[1-p]) ----
    // Safe without a second barrier: acc[1-p] was fully written before this
    // step's barrier; the next writer of acc[1-p] is B(t+1), which is behind
    // the barrier wave 0 only signals after finishing this read.
    if (t > 0 && tid < ROWS) {
      const float* a = &accbuf[1 - p][tid * NWAVES];
      v4f a0 = *(const v4f*)a;
      v4f a1 = *(const v4f*)(a + 4);
      float s = (a0[0] + a0[1]) + (a0[2] + a0[3]) + ((a1[0] + a1[1]) + (a1[2] + a1[3]));
      m3 = __builtin_fmaf(c_b3, m3, s + c_c3);
      out[(t - 1) * BATCH + bbase + tid] = m3;
    }
  }

  // drain: output for the final timestep (t = T_STEPS-1, parity 1)
  __syncthreads();
  if (tid < ROWS) {
    const float* a = &accbuf[(T_STEPS - 1) & 1][tid * NWAVES];
    v4f a0 = *(const v4f*)a;
    v4f a1 = *(const v4f*)(a + 4);
    float s = (a0[0] + a0[1]) + (a0[2] + a0[3]) + ((a1[0] + a1[1]) + (a1[2] + a1[3]));
    m3 = __builtin_fmaf(c_b3, m3, s + c_c3);
    out[(T_STEPS - 1) * BATCH + bbase + tid] = m3;
  }
}

extern "C" void kernel_launch(void* const* d_in, const int* in_sizes, int n_in,
                              void* d_out, int out_size, void* d_ws, size_t ws_size,
                              hipStream_t stream) {
  const float* x        = (const float*)d_in[0];
  const float* W_in     = (const float*)d_in[1];
  const float* b_in     = (const float*)d_in[2];
  const float* beta_in  = (const float*)d_in[3];
  const float* thr_in   = (const float*)d_in[4];
  const float* W_h      = (const float*)d_in[5];
  const float* b_h      = (const float*)d_in[6];
  const float* beta_h   = (const float*)d_in[7];
  const float* thr_h    = (const float*)d_in[8];
  const float* W_out    = (const float*)d_in[9];
  const float* b_out    = (const float*)d_in[10];
  const float* beta_out = (const float*)d_in[11];
  float* out            = (float*)d_out;

  dim3 grid(BATCH / ROWS);   // 128 workgroups, each owns 16 batch rows
  dim3 block(256);           // 8 waves (wave32)
  hipLaunchKernelGGL(snn_scan_kernel, grid, block, 0, stream,
                     x, W_in, b_in, beta_in, thr_in, W_h, b_h, beta_h, thr_h,
                     W_out, b_out, beta_out, out);
}
